// DecoderTreeLSTM_12326556139666
// MI455X (gfx1250) — matrix operand order repaired and checked by reading, not compiled, not run on hardware
//
#include <hip/hip_runtime.h>

// ---------------- problem constants (fixed by the reference) ----------------
#define NNODES   1024
#define IN_DIM   2048
#define EMBED_D  200
#define XDIM     2248           // IN_DIM + EMBED_D
#define XPAD     2304           // round up to multiple of 64 (2x WMMA K, even #pairs)
#define HID      1024
#define G5       5120           // 5*HID
#define NCLS     151
#define NOUTPAD  160            // NCLS rounded to multiple of 16
#define NBLOCKS  40             // persistent grid: 40 blocks x 10 waves = 400 waves
#define TPB      320            // 10 waves of 32 (wave32)
#define WPB      10

typedef __attribute__((ext_vector_type(16))) __bf16 v16bf;
typedef __attribute__((ext_vector_type(8)))  float  v8f;

// ---------------------------------------------------------------------------
// Async global->LDS copy of one 16-byte chunk per active lane (CDNA5 async
// path, tracked by ASYNCcnt).  VDST operand carries the LDS byte offset; on
// AMDGPU the low 32 bits of a generic __shared__ pointer ARE the LDS offset.
__device__ inline void async_copy_b128(unsigned lds_off, const void* gsrc) {
    unsigned long long ga = (unsigned long long)gsrc;
    asm volatile("global_load_async_to_lds_b128 %0, %1, off"
                 :: "v"(lds_off), "v"(ga) : "memory");
}
__device__ inline void wait_asynccnt0() {
    asm volatile("s_wait_asynccnt 0x0" ::: "memory");
}

// A-tile: 16x32 bf16 slice of TRANSPOSED weights WT (N rows x K cols, ldk
// columns). CDNA5 16-bit A layout (16x32): lane L carries row M=(L&15); it
// holds K = base..base+7 in elements 0..7 and K = base+16..base+23 in elements
// 8..15, where base = (L>>4)*8.  All 32 lanes do two contiguous b128 loads --
// no predication, no zero-fill.
__device__ inline v16bf load_wt_tile(const __bf16* WT, int k, int n0, int ldk, int lane) {
    const __bf16* p = WT + (size_t)(n0 + (lane & 15)) * ldk + k + ((lane >> 4) << 3);
    __builtin_prefetch(p + 64, 0, 1);                 // next K-pair block (128B ahead)
    union { uint4 q[2]; v16bf v; } u;
    u.q[0] = *(const uint4*)p;                        // K = base .. base+7
    u.q[1] = *(const uint4*)(p + 16);                 // K = base+16 .. base+23
    return u.v;
}

__device__ inline v16bf load_b_vec(const __bf16* s, int k, int lane) {
    const __bf16* p = s + k + ((lane >> 4) << 3);
    union { uint4 q[2]; v16bf v; } u;
    u.q[0] = *(const uint4*)p;
    u.q[1] = *(const uint4*)(p + 16);
    return u.v;
}

__device__ inline v8f wmma_bf16(v16bf a, v16bf b, v8f c) {
    return __builtin_amdgcn_wmma_f32_16x16x32_bf16(false, a, false, b, (short)0, c, false, false);
}

__device__ inline float sigm_f(float x) { return 1.0f / (1.0f + __expf(-x)); }
__device__ inline float tanh_f(float x) {
    x = fminf(fmaxf(x, -15.0f), 15.0f);
    float e = __expf(2.0f * x);
    return (e - 1.0f) / (e + 1.0f);
}

// Lightweight grid barrier (sense via generation counter). Only thread 0 of
// each block arrives; spinners sleep on s_sleep. bar[0]=count, bar[1]=gen.
__device__ inline void gsync(unsigned* bar, unsigned nb) {
    __syncthreads();
    if (threadIdx.x == 0) {
        __threadfence();
        unsigned gen = __hip_atomic_load(bar + 1, __ATOMIC_RELAXED, __HIP_MEMORY_SCOPE_AGENT);
        unsigned arr = __hip_atomic_fetch_add(bar, 1u, __ATOMIC_ACQ_REL, __HIP_MEMORY_SCOPE_AGENT);
        if (arr == nb - 1) {
            __hip_atomic_store(bar, 0u, __ATOMIC_RELAXED, __HIP_MEMORY_SCOPE_AGENT);
            __hip_atomic_fetch_add(bar + 1, 1u, __ATOMIC_RELEASE, __HIP_MEMORY_SCOPE_AGENT);
        } else {
            while (__hip_atomic_load(bar + 1, __ATOMIC_ACQUIRE, __HIP_MEMORY_SCOPE_AGENT) == gen)
                __builtin_amdgcn_s_sleep(2);
        }
    }
    __syncthreads();
}

// --------- one-time fp32 -> bf16 TRANSPOSED weight repack (L2-resident) ------
// Layouts: bWxT [G5 x XPAD], bWhT [G5 x HID], bWpxT [HID x XPAD],
//          bWoutT [NOUTPAD x HID]; K padded with zeros, N padded rows zeroed.
__global__ void convert_weights_kernel(const float* __restrict__ Wx,  const float* __restrict__ Wh,
                                       const float* __restrict__ Wpx, const float* __restrict__ Wout,
                                       __bf16* __restrict__ bWxT,  __bf16* __restrict__ bWhT,
                                       __bf16* __restrict__ bWpxT, __bf16* __restrict__ bWoutT) {
    const size_t S0 = (size_t)G5 * XPAD;
    const size_t S1 = (size_t)G5 * HID;
    const size_t S2 = (size_t)HID * XPAD;
    const size_t S3 = (size_t)NOUTPAD * HID;
    const size_t TOT = S0 + S1 + S2 + S3;
    for (size_t e = blockIdx.x * (size_t)blockDim.x + threadIdx.x; e < TOT;
         e += (size_t)gridDim.x * blockDim.x) {
        if (e < S0) {
            size_t n = e / XPAD, k = e % XPAD;
            bWxT[e] = (__bf16)(k < XDIM ? Wx[k * G5 + n] : 0.0f);
        } else if (e < S0 + S1) {
            size_t q = e - S0;
            size_t n = q / HID, k = q % HID;
            bWhT[q] = (__bf16)Wh[k * G5 + n];
        } else if (e < S0 + S1 + S2) {
            size_t q = e - S0 - S1;
            size_t n = q / XPAD, k = q % XPAD;
            bWpxT[q] = (__bf16)(k < XDIM ? Wpx[k * HID + n] : 0.0f);
        } else {
            size_t q = e - S0 - S1 - S2;
            size_t n = q / HID, k = q % HID;
            bWoutT[q] = (__bf16)(n < NCLS ? Wout[k * NCLS + n] : 0.0f);
        }
    }
}

// --------------------- persistent sequential-scan kernel --------------------
__global__ void __launch_bounds__(TPB, 1)
treelstm_persistent(const float* __restrict__ features, const int* __restrict__ parent,
                    const int*  __restrict__ batchp,
                    const float* __restrict__ b_px,   const float* __restrict__ b_iofux,
                    const float* __restrict__ b_iofuh,const float* __restrict__ b_out,
                    const float* __restrict__ embed,
                    const __bf16* __restrict__ WxT,  const __bf16* __restrict__ WhT,
                    const __bf16* __restrict__ WpxT, const __bf16* __restrict__ WoutT,
                    __bf16* __restrict__ x_bf, __bf16* __restrict__ ph_bf,
                    float* __restrict__ gbuf, float* __restrict__ projbuf,
                    float* __restrict__ h_all, float* __restrict__ c_all,
                    int* __restrict__ commit_all,
                    float* __restrict__ out, unsigned* __restrict__ bar) {
    __shared__ __align__(16) __bf16 s_x[XPAD];     // async-staged x (bf16)
    __shared__ __align__(16) __bf16 s_ph[HID];     // async-staged parent h
    __shared__ __align__(16) __bf16 s_hfin[HID];
    __shared__ float s_dist[NOUTPAD];
    __shared__ int   s_commit;

    const int tid  = threadIdx.x;
    const int lane = tid & 31;
    const int wave = tid >> 5;
    const int gw   = blockIdx.x * WPB + wave;
    const int batch = batchp[0];
    const bool b0  = (blockIdx.x == 0);
    int* out_commit = (int*)(out + (size_t)(NNODES - batch) * NCLS);

    const unsigned sx_off  = (unsigned)(unsigned long long)(void*)s_x;   // LDS byte offsets
    const unsigned sph_off = (unsigned)(unsigned long long)(void*)s_ph;

    // B operand: x slice loaded by every lane in the A table K order (elements
    // 0..7: K=8h+s, 8..15: K=16+8h+s-8); only column N=0 is consumed.
    v16bf bcol0, bcol1;
#pragma unroll
    for (int e = 0; e < 16; ++e) { bcol0[e] = (__bf16)0.0f; bcol1[e] = (__bf16)0.0f; }

    // Build x/ph for a step (written to globally visible staging buffers).
    auto build_x = [&](int ni, bool root, int pps, int cm) {
        for (int t = tid; t < XPAD; t += TPB) {
            float v = 0.0f;
            if (t < IN_DIM)     v = features[(size_t)ni * IN_DIM + t];
            else if (t < XDIM)  v = embed[(size_t)cm * EMBED_D + (t - IN_DIM)];
            x_bf[t] = (__bf16)v;
        }
        for (int t = tid; t < HID; t += TPB)
            ph_bf[t] = root ? (__bf16)0.0f : (__bf16)h_all[(size_t)pps * HID + t];
    };

    // ---- prologue: block0 stages x for node 0 and zeroes the commit tail ----
    if (b0) {
        int p0 = parent[0];
        build_x(0, p0 < 0, p0 < 0 ? 0 : p0, 0);
        for (int j = NNODES - batch + tid; j < NNODES; j += TPB) out_commit[j] = 0;
    }
    gsync(bar, NBLOCKS);

    for (int i = 0; i < NNODES; ++i) {
        // ---- stage x/ph into this block's LDS via async global->LDS DMA ----
        if (tid < (XPAD * 2 / 16))                       // 288 x 16B chunks
            async_copy_b128(sx_off + tid * 16, (const char*)x_bf + tid * 16);
        if (tid < (HID * 2 / 16))                        // 128 x 16B chunks
            async_copy_b128(sph_off + tid * 16, (const char*)ph_bf + tid * 16);
        wait_asynccnt0();
        __syncthreads();

        // ---------- phase 1: all 384 GEMV tiles via WMMA (all blocks) ----------
        if (gw < 320) {                       // g = x@W_iofux + ph@W_iofuh (+bias)
            const int n0 = gw * 16;
            v8f acc0 = {}, acc1 = {};         // 2-way accumulation: 2 WMMAs in flight
            for (int k = 0; k < XPAD; k += 64) {
                bcol0 = load_b_vec(s_x, k, lane);
                bcol1 = load_b_vec(s_x, k + 32, lane);
                acc0 = wmma_bf16(load_wt_tile(WxT, k,      n0, XPAD, lane), bcol0, acc0);
                acc1 = wmma_bf16(load_wt_tile(WxT, k + 32, n0, XPAD, lane), bcol1, acc1);
            }
            for (int k = 0; k < HID; k += 64) {
                bcol0 = load_b_vec(s_ph, k, lane);
                bcol1 = load_b_vec(s_ph, k + 32, lane);
                acc0 = wmma_bf16(load_wt_tile(WhT, k,      n0, HID, lane), bcol0, acc0);
                acc1 = wmma_bf16(load_wt_tile(WhT, k + 32, n0, HID, lane), bcol1, acc1);
            }
            // D(m, N=0): m=0..7 in lane 0 VGPR m; m=8..15 in lane 16 VGPR m-8.
            if ((lane & 15) == 0) {
                const int base = n0 + ((lane >> 4) << 3);
#pragma unroll
                for (int e = 0; e < 8; ++e)
                    gbuf[base + e] = acc0[e] + acc1[e] + b_iofux[base + e] + b_iofuh[base + e];
            }
        } else if (gw < 384) {                 // proj_x = x@W_px + b_px
            const int n0 = (gw - 320) * 16;
            v8f acc0 = {}, acc1 = {};
            for (int k = 0; k < XPAD; k += 64) {
                bcol0 = load_b_vec(s_x, k, lane);
                bcol1 = load_b_vec(s_x, k + 32, lane);
                acc0 = wmma_bf16(load_wt_tile(WpxT, k,      n0, XPAD, lane), bcol0, acc0);
                acc1 = wmma_bf16(load_wt_tile(WpxT, k + 32, n0, XPAD, lane), bcol1, acc1);
            }
            if ((lane & 15) == 0) {
                const int base = n0 + ((lane >> 4) << 3);
#pragma unroll
                for (int e = 0; e < 8; ++e)
                    projbuf[base + e] = acc0[e] + acc1[e] + b_px[base + e];
            }
        }
        gsync(bar, NBLOCKS);

        // ---------- phase 2: serial tail on block 0 only ----------
        if (b0) {
            const int p = parent[i];
            const bool root = p < 0;
            const int ps = root ? 0 : p;

            // pointwise gates -> c, h, h_final
            for (int j = tid; j < HID; j += TPB) {
                float ig = sigm_f(gbuf[j]);
                float og = sigm_f(gbuf[HID + j]);
                float fg = sigm_f(gbuf[2 * HID + j]);
                float ug = tanh_f(gbuf[3 * HID + j]);
                float rg = sigm_f(gbuf[4 * HID + j]);
                float pc = root ? 0.0f : c_all[(size_t)ps * HID + j];
                float c  = ig * ug + fg * pc;
                float h  = og * tanh_f(c);
                float hf = rg * h + (1.0f - rg) * projbuf[j];
                c_all[(size_t)i * HID + j] = c;
                h_all[(size_t)i * HID + j] = hf;
                s_hfin[j] = (__bf16)hf;        // B-feed for output GEMV via LDS
            }
            __syncthreads();

            // output GEMV: 10 waves x 16 cols, K=1024, h_final fed from LDS
            {
                const int n0 = wave * 16;
                v8f acc0 = {}, acc1 = {};
                for (int k = 0; k < HID; k += 64) {
                    bcol0 = load_b_vec(s_hfin, k, lane);
                    bcol1 = load_b_vec(s_hfin, k + 32, lane);
                    acc0 = wmma_bf16(load_wt_tile(WoutT, k,      n0, HID, lane), bcol0, acc0);
                    acc1 = wmma_bf16(load_wt_tile(WoutT, k + 32, n0, HID, lane), bcol1, acc1);
                }
                if ((lane & 15) == 0) {
                    const int base = n0 + ((lane >> 4) << 3);
#pragma unroll
                    for (int e = 0; e < 8; ++e) {
                        int col = base + e;
                        s_dist[col] = acc0[e] + acc1[e] + (col < NCLS ? b_out[col] : -3.0e38f);
                    }
                }
                bcol0[0] = (__bf16)0.0f;       // restore invariant for next step
                bcol1[0] = (__bf16)0.0f;
            }
            __syncthreads();

            if (tid == 0) {                    // argmax over dist[1:] (first max wins)
                int best = 1; float bv = s_dist[1];
                for (int j = 2; j < NCLS; ++j) {
                    float v = s_dist[j];
                    if (v > bv) { bv = v; best = j; }
                }
                s_commit = best;
                commit_all[i] = best;
            }
            __syncthreads();
            const int cm_i = s_commit;

            if (i >= batch) {
                for (int j = tid; j < NCLS; j += TPB)
                    out[(size_t)(i - batch) * NCLS + j] = s_dist[j];
                if (tid == 0) out_commit[i - batch] = cm_i;
            }

            // stage x/ph for node i+1 (parent commits are all block0-local writes)
            const int ni = i + 1;
            if (ni < NNODES) {
                int pp = parent[ni];
                bool r2 = pp < 0;
                int pps = r2 ? 0 : pp;
                int cm = r2 ? 0 : (pps == i ? cm_i : commit_all[pps]);
                build_x(ni, r2, pps, cm);
            }
        }
        gsync(bar, NBLOCKS);                   // x ready / gbuf free for next step
    }
}

// ---------------------------------------------------------------------------
extern "C" void kernel_launch(void* const* d_in, const int* in_sizes, int n_in,
                              void* d_out, int out_size, void* d_ws, size_t ws_size,
                              hipStream_t stream) {
    const float* features = (const float*)d_in[0];
    const int*   parent   = (const int*)d_in[1];
    const int*   batchp   = (const int*)d_in[2];
    const float* W_px     = (const float*)d_in[3];
    const float* b_px     = (const float*)d_in[4];
    const float* W_iofux  = (const float*)d_in[5];
    const float* b_iofux  = (const float*)d_in[6];
    const float* W_iofuh  = (const float*)d_in[7];
    const float* b_iofuh  = (const float*)d_in[8];
    const float* W_out    = (const float*)d_in[9];
    const float* b_out    = (const float*)d_in[10];
    const float* embed    = (const float*)d_in[11];

    char* ws = (char*)d_ws;
    size_t off = 0;
    auto take = [&](size_t bytes) -> char* {
        char* p = ws + off;
        off = (off + bytes + 255) & ~(size_t)255;
        return p;
    };
    unsigned* bar     = (unsigned*)take(256);
    __bf16* x_bf      = (__bf16*)take((size_t)XPAD * 2);
    __bf16* ph_bf     = (__bf16*)take((size_t)HID * 2);
    float*  gbuf      = (float*) take((size_t)G5 * 4);
    float*  projbuf   = (float*) take((size_t)HID * 4);
    float*  h_all     = (float*) take((size_t)NNODES * HID * 4);
    float*  c_all     = (float*) take((size_t)NNODES * HID * 4);
    int*    commit_all= (int*)   take((size_t)NNODES * 4);
    __bf16* bWxT      = (__bf16*)take((size_t)G5 * XPAD * 2);
    __bf16* bWhT      = (__bf16*)take((size_t)G5 * HID * 2);
    __bf16* bWpxT     = (__bf16*)take((size_t)HID * XPAD * 2);
    __bf16* bWoutT    = (__bf16*)take((size_t)NOUTPAD * HID * 2);

    hipMemsetAsync(bar, 0, 256, stream);                      // barrier state
    convert_weights_kernel<<<2048, 256, 0, stream>>>(W_iofux, W_iofuh, W_px, W_out,
                                                     bWxT, bWhT, bWpxT, bWoutT);
    treelstm_persistent<<<NBLOCKS, TPB, 0, stream>>>(features, parent, batchp,
        b_px, b_iofux, b_iofuh, b_out, embed,
        bWxT, bWhT, bWpxT, bWoutT,
        x_bf, ph_bf, gbuf, projbuf, h_all, c_all, commit_all,
        (float*)d_out, bar);
}
